// FPSPointNetSetAbstractionMsg_12506944766650
// MI455X (gfx1250) — matrix operand order repaired and checked by reading, not compiled, not run on hardware
//
#include <hip/hip_runtime.h>

// ---------------------------------------------------------------------------
// FPSPointNetSetAbstractionMsg for MI455X (gfx1250, wave32, WMMA bf16)
// ---------------------------------------------------------------------------
// Pipeline:
//   1) pack_kernel      : fold BN into per-channel scale/shift, W -> bf16 padded
//   2) fps_kernel       : farthest point sampling, dist array in LDS
//   3) ballq_kernel     : ball query via wave32 ballot (index-ordered, matches ref)
//   4) gather_kernel    : build X[col][96] bf16 (64 feat + 3 rel-xyz + zero pad)
//   5) gemm_bn_relu_wmma: D = relu(fma(W@X, scale, shift)) via v_wmma_f32_16x16x32_bf16
//   6) maxpool_kernel   : max over K -> new_points output
//   7) refine_offset    : softmax over K (wave reduction) + weighted xyz sum
// Layout choice: activations are [col][ch_pad] so both WMMA operands and the
// result are contiguous 128-bit vmem accesses (no transpose loads needed).
// ---------------------------------------------------------------------------

#define BB   8
#define NN   8192
#define SS   512

typedef __bf16 bf16;
typedef __attribute__((ext_vector_type(16))) __bf16 v16bf;
typedef __attribute__((ext_vector_type(8)))  __bf16 v8bf;
typedef __attribute__((ext_vector_type(8)))  float  v8f;

// --------------------------- weight packing --------------------------------
__global__ void pack_kernel(const float* __restrict__ W, const float* __restrict__ b,
                            const float* __restrict__ g, const float* __restrict__ be,
                            const float* __restrict__ mn, const float* __restrict__ vr,
                            bf16* __restrict__ Wp, float* __restrict__ scale,
                            float* __restrict__ shift,
                            int cout, int cin, int coutp, int cinp)
{
    int i = blockIdx.x * blockDim.x + threadIdx.x;
    int tot = coutp * cinp;
    if (i < tot) {
        int r = i / cinp, c = i % cinp;
        float w = (r < cout && c < cin) ? W[r * cin + c] : 0.f;
        Wp[i] = (bf16)w;
    }
    if (i < coutp) {
        if (i < cout) {
            float s = g[i] * rsqrtf(vr[i] + 1e-5f);
            scale[i] = s;
            shift[i] = (b[i] - mn[i]) * s + be[i];
        } else {
            scale[i] = 0.f;
            shift[i] = 0.f;
        }
    }
}

// --------------------------- farthest point sampling -----------------------
__global__ __launch_bounds__(1024) void fps_kernel(const float* __restrict__ xyz,
                                                   int* __restrict__ fpsIdx,
                                                   float* __restrict__ nxWs,
                                                   float* __restrict__ out0)
{
    const int b = blockIdx.x;
    const float* xb = xyz + (size_t)b * 3 * NN;
    __shared__ float dist[NN];          // 32 KB of the 320 KB WGP LDS
    __shared__ float rv[32];
    __shared__ int   ri[32];
    __shared__ int   sFar;
    const int t = threadIdx.x;

    for (int p = t; p < NN; p += 1024) dist[p] = 1e10f;
    if (t == 0) sFar = 0;
    __syncthreads();

    for (int it = 0; it < SS; ++it) {
        const int far = sFar;
        const float cx = xb[far], cy = xb[NN + far], cz = xb[2 * NN + far];
        if (t == 0) {
            fpsIdx[b * SS + it] = far;
            nxWs[(b * SS + it) * 3 + 0] = cx;
            nxWs[(b * SS + it) * 3 + 1] = cy;
            nxWs[(b * SS + it) * 3 + 2] = cz;
            out0[(size_t)b * 3 * SS + 0 * SS + it] = cx;
            out0[(size_t)b * 3 * SS + 1 * SS + it] = cy;
            out0[(size_t)b * 3 * SS + 2 * SS + it] = cz;
        }
        float bv = -1e30f; int bi = 0;
        for (int p = t; p < NN; p += 1024) {
            float dx = xb[p] - cx, dy = xb[NN + p] - cy, dz = xb[2 * NN + p] - cz;
            float d = dx * dx + dy * dy + dz * dz;
            float dd = fminf(dist[p], d);
            dist[p] = dd;
            if (dd > bv) { bv = dd; bi = p; }
        }
        // wave32 argmax reduction (prefer lower index on tie, matches argmax)
        for (int off = 16; off; off >>= 1) {
            float ov = __shfl_xor(bv, off);
            int   oi = __shfl_xor(bi, off);
            if (ov > bv || (ov == bv && oi < bi)) { bv = ov; bi = oi; }
        }
        __syncthreads();
        if ((t & 31) == 0) { rv[t >> 5] = bv; ri[t >> 5] = bi; }
        __syncthreads();
        if (t == 0) {
            float best = rv[0]; int bidx = ri[0];
            for (int w = 1; w < 32; ++w)
                if (rv[w] > best || (rv[w] == best && ri[w] < bidx)) { best = rv[w]; bidx = ri[w]; }
            sFar = bidx;
        }
        __syncthreads();
    }
}

// --------------------------- ball query ------------------------------------
__global__ void ballq_kernel(const float* __restrict__ xb, const float* __restrict__ nx,
                             int* __restrict__ idxbuf, float r2, int K)
{
    const int wv   = (blockIdx.x * blockDim.x + threadIdx.x) >> 5;
    const int lane = threadIdx.x & 31;
    if (wv >= SS) return;
    const float cx = nx[wv * 3 + 0], cy = nx[wv * 3 + 1], cz = nx[wv * 3 + 2];
    int cnt = 0, firstIdx = 0, haveFirst = 0;
    for (int base = 0; base < NN && cnt < K; base += 32) {
        const int p = base + lane;
        float dx = xb[p] - cx, dy = xb[NN + p] - cy, dz = xb[2 * NN + p] - cz;
        bool in = (dx * dx + dy * dy + dz * dz) <= r2;
        unsigned mask = (unsigned)__ballot(in);
        int pre = __popc(mask & ((1u << lane) - 1u));
        if (in && cnt + pre < K) idxbuf[wv * K + cnt + pre] = p;
        if (!haveFirst && mask) { firstIdx = base + (__ffs(mask) - 1); haveFirst = 1; }
        cnt += __popc(mask);
    }
    if (cnt < K) {
        int fill = haveFirst ? firstIdx : 0;
        for (int j = cnt + lane; j < K; j += 32) idxbuf[wv * K + j] = fill;
    }
}

// --------------------------- grouped feature gather ------------------------
__global__ void gather_kernel(const float* __restrict__ xyzb, const float* __restrict__ ptsb,
                              const int* __restrict__ idx, const float* __restrict__ nxb,
                              bf16* __restrict__ X, float* __restrict__ gx,
                              int K, int writeGx)
{
    int n = blockIdx.x * blockDim.x + threadIdx.x;
    if (n >= SS * K) return;
    int s = n / K;
    int p = idx[n];
    bf16* xr = X + (size_t)n * 96;
#pragma unroll 8
    for (int c = 0; c < 64; ++c) xr[c] = (bf16)ptsb[(size_t)c * NN + p];
    float dx = xyzb[p]          - nxb[s * 3 + 0];
    float dy = xyzb[NN + p]     - nxb[s * 3 + 1];
    float dz = xyzb[2 * NN + p] - nxb[s * 3 + 2];
    xr[64] = (bf16)dx; xr[65] = (bf16)dy; xr[66] = (bf16)dz;
#pragma unroll
    for (int c = 67; c < 96; ++c) xr[c] = (bf16)0.f;
    if (writeGx) { gx[n * 3 + 0] = dx; gx[n * 3 + 1] = dy; gx[n * 3 + 2] = dz; }
}

// --------------------------- WMMA GEMM + BN + ReLU -------------------------
// Y[n][coutp] = relu(fma(sum_k W[m][k]*X[n][k], scale[m], shift[m]))
// Wave computes a 16(M) x 64(N) tile; block = 8 waves = 16x512; exact tiling.
__global__ __launch_bounds__(256) void gemm_bn_relu_wmma(
    const bf16* __restrict__ X, const bf16* __restrict__ Wp,
    const float* __restrict__ scale, const float* __restrict__ shift,
    bf16* __restrict__ Y, int Ncols, int cinp, int coutp)
{
    const int lane    = threadIdx.x & 31;
    const int wave    = threadIdx.x >> 5;
    const int l15     = lane & 15;
    const int khalf   = lane >> 4;                 // 0/1 per ISA A/B lane split
    const int mt      = blockIdx.y;
    const int colBase = blockIdx.x * 512 + wave * 64;
    const int mRow    = mt * 16 + l15;
    (void)Ncols;

    v8f acc0 = {0.f,0.f,0.f,0.f,0.f,0.f,0.f,0.f};
    v8f acc1 = acc0, acc2 = acc0, acc3 = acc0;

    const bf16* Wrow = Wp + (size_t)mRow * cinp;
    const bf16* Xcol = X + (size_t)(colBase + l15) * cinp + khalf * 16;

    for (int kt = 0; kt < cinp; kt += 32) {
        // A: 16x32 bf16. lane<16: row m, K {0..15}; lane>=16: K {16..31}
        v8bf a0 = *(const v8bf*)(Wrow + kt + khalf * 16);
        v8bf a1 = *(const v8bf*)(Wrow + kt + khalf * 16 + 8);
        v16bf a;
#pragma unroll
        for (int i = 0; i < 8; ++i) { a[i] = a0[i]; a[i + 8] = a1[i]; }

        // B: 32x16 bf16 per tile. lane n fixed, 16 consecutive k -> 32B load
        v16bf b0 = *(const v16bf*)(Xcol + kt + (size_t)0 * 16 * cinp);
        v16bf b1 = *(const v16bf*)(Xcol + kt + (size_t)1 * 16 * cinp);
        v16bf b2 = *(const v16bf*)(Xcol + kt + (size_t)2 * 16 * cinp);
        v16bf b3 = *(const v16bf*)(Xcol + kt + (size_t)3 * 16 * cinp);

        __builtin_prefetch((const void*)(Xcol + kt + 32), 0, 1);

        acc0 = __builtin_amdgcn_wmma_f32_16x16x32_bf16(false, a, false, b0, (short)0, acc0, false, false);
        acc1 = __builtin_amdgcn_wmma_f32_16x16x32_bf16(false, a, false, b1, (short)0, acc1, false, false);
        acc2 = __builtin_amdgcn_wmma_f32_16x16x32_bf16(false, a, false, b2, (short)0, acc2, false, false);
        acc3 = __builtin_amdgcn_wmma_f32_16x16x32_bf16(false, a, false, b3, (short)0, acc3, false, false);
    }

    // C/D layout: lane<16 -> rows v, lane>=16 -> rows 8+v; n = l15 (+ j*16)
    const int mBase = mt * 16 + khalf * 8;
    float sc[8], sh[8];
#pragma unroll
    for (int v = 0; v < 8; ++v) { sc[v] = scale[mBase + v]; sh[v] = shift[mBase + v]; }

    v8f accs[4] = {acc0, acc1, acc2, acc3};
#pragma unroll
    for (int j = 0; j < 4; ++j) {
        const int n = colBase + j * 16 + l15;
        v8bf o;
#pragma unroll
        for (int v = 0; v < 8; ++v) {
            float t = fmaf(accs[j][v], sc[v], sh[v]);
            t = t > 0.f ? t : 0.f;
            o[v] = (bf16)t;
        }
        *(v8bf*)(Y + (size_t)n * coutp + mBase) = o;  // 16B packed bf16 store
    }
}

// --------------------------- max pool over K -------------------------------
__global__ void maxpool_kernel(const bf16* __restrict__ act, float* __restrict__ out2,
                               int K, int coutp, int cout)
{
    int i = blockIdx.x * blockDim.x + threadIdx.x;
    if (i >= cout * SS) return;
    int c = i / SS, s = i % SS;
    float m = -1e30f;
    for (int k = 0; k < K; ++k)
        m = fmaxf(m, (float)act[((size_t)(s * K + k)) * coutp + c]);
    out2[c * SS + s] = m;
}

// --------------------------- softmax(K) * grouped_xyz ----------------------
__global__ void refine_offset_kernel(const bf16* __restrict__ act,
                                     const float* __restrict__ gx,
                                     float* __restrict__ out1)
{
    const int wv   = (blockIdx.x * blockDim.x + threadIdx.x) >> 5;
    const int lane = threadIdx.x & 31;
    if (wv >= SS) return;
    const int s = wv;
    float v[4], mx = -1e30f;
#pragma unroll
    for (int j = 0; j < 4; ++j) {
        int k = j * 32 + lane;
        v[j] = (float)act[((size_t)(s * 128 + k)) * 32];   // refine output ch 0
        mx = fmaxf(mx, v[j]);
    }
    for (int off = 16; off; off >>= 1) mx = fmaxf(mx, __shfl_xor(mx, off));
    float sum = 0.f, ox = 0.f, oy = 0.f, oz = 0.f;
#pragma unroll
    for (int j = 0; j < 4; ++j) {
        int k = j * 32 + lane;
        float e = __expf(v[j] - mx);
        sum += e;
        const float* g = gx + (size_t)(s * 128 + k) * 3;
        ox += e * g[0]; oy += e * g[1]; oz += e * g[2];
    }
    for (int off = 16; off; off >>= 1) {
        sum += __shfl_xor(sum, off);
        ox  += __shfl_xor(ox, off);
        oy  += __shfl_xor(oy, off);
        oz  += __shfl_xor(oz, off);
    }
    if (lane == 0) {
        float inv = 1.f / sum;
        out1[0 * SS + s] = ox * inv;
        out1[1 * SS + s] = oy * inv;
        out1[2 * SS + s] = oz * inv;
    }
}

// --------------------------- host orchestration ----------------------------
extern "C" void kernel_launch(void* const* d_in, const int* in_sizes, int n_in,
                              void* d_out, int out_size, void* d_ws, size_t ws_size,
                              hipStream_t stream)
{
    (void)in_sizes; (void)n_in; (void)out_size; (void)ws_size;
    const float* xyz = (const float*)d_in[0];
    const float* pts = (const float*)d_in[1];
    float* out = (float*)d_out;
    char* ws = (char*)d_ws;

    // 9 MSG-MLP layers then 11 refine layers; d_in base for layer L is 2+6L
    static const int LC[20][2] = {
        {64,67},{64,64},{128,64},{128,67},{128,128},{256,128},{128,67},{128,128},{256,128},
        {512,67},{256,512},{128,256},{64,128},{32,64},{16,32},{8,16},{4,8},{2,4},{1,2},{1,1}};
    int coutp[20], cinp[20];
    size_t offW[20], offSc[20], offSh[20];
    size_t off = 0;
    auto take = [&](size_t bytes) { size_t o = off; off += (bytes + 255) & ~(size_t)255; return o; };
    size_t offFps = take((size_t)BB * SS * 4);
    size_t offNx  = take((size_t)BB * SS * 3 * 4);
    for (int L = 0; L < 20; ++L) {
        coutp[L] = (LC[L][0] + 31) & ~31;
        cinp[L]  = (LC[L][1] + 31) & ~31;
        offW[L]  = take((size_t)coutp[L] * cinp[L] * 2);
        offSc[L] = take((size_t)coutp[L] * 4);
        offSh[L] = take((size_t)coutp[L] * 4);
    }
    size_t offIdx = take((size_t)65536 * 4);
    size_t offX   = take((size_t)65536 * 96 * 2);
    size_t offGx  = take((size_t)65536 * 3 * 4);
    size_t offA   = take((size_t)65536 * 512 * 2);
    size_t offB2  = take((size_t)65536 * 512 * 2);

    // 1) fold BN + pack weights to padded bf16
    for (int L = 0; L < 20; ++L) {
        const float* W  = (const float*)d_in[2 + 6 * L + 0];
        const float* bb = (const float*)d_in[2 + 6 * L + 1];
        const float* gg = (const float*)d_in[2 + 6 * L + 2];
        const float* be = (const float*)d_in[2 + 6 * L + 3];
        const float* mn = (const float*)d_in[2 + 6 * L + 4];
        const float* vr = (const float*)d_in[2 + 6 * L + 5];
        int tot = coutp[L] * cinp[L];
        pack_kernel<<<(tot + 255) / 256, 256, 0, stream>>>(
            W, bb, gg, be, mn, vr, (bf16*)(ws + offW[L]),
            (float*)(ws + offSc[L]), (float*)(ws + offSh[L]),
            LC[L][0], LC[L][1], coutp[L], cinp[L]);
    }

    // 2) FPS (also writes new_xyz output)
    fps_kernel<<<BB, 1024, 0, stream>>>(xyz, (int*)(ws + offFps), (float*)(ws + offNx), out);

    const int   Ks[3]    = {16, 32, 128};
    const float R2[3]    = {0.01f, 0.04f, 0.16f};
    const int   chOff[3] = {0, 128, 384};

    for (int b = 0; b < BB; ++b) {
        const float* xb = xyz + (size_t)b * 3 * NN;
        const float* pb = pts + (size_t)b * 64 * NN;
        const float* nx = (const float*)(ws + offNx) + (size_t)b * SS * 3;
        for (int br = 0; br < 3; ++br) {
            const int K = Ks[br];
            const int Ncols = K * SS;
            int* idxb = (int*)(ws + offIdx);

            ballq_kernel<<<(SS * 32) / 128, 128, 0, stream>>>(xb, nx, idxb, R2[br], K);
            gather_kernel<<<(Ncols + 255) / 256, 256, 0, stream>>>(
                xb, pb, idxb, nx, (bf16*)(ws + offX), (float*)(ws + offGx), K, br == 2 ? 1 : 0);

            // MSG MLP chain (3 layers)
            const bf16* src = (const bf16*)(ws + offX);
            bf16* bufs[2] = {(bf16*)(ws + offA), (bf16*)(ws + offB2)};
            for (int j = 0; j < 3; ++j) {
                int L = br * 3 + j;
                bf16* o = bufs[j & 1];
                dim3 g(Ncols / 512, coutp[L] / 16);
                gemm_bn_relu_wmma<<<g, 256, 0, stream>>>(
                    src, (const bf16*)(ws + offW[L]),
                    (const float*)(ws + offSc[L]), (const float*)(ws + offSh[L]),
                    o, Ncols, cinp[L], coutp[L]);
                src = o;
            }
            int Llast = br * 3 + 2;
            float* out2 = out + 2 * BB * 3 * SS + (size_t)b * 640 * SS + (size_t)chOff[br] * SS;
            int tot = LC[Llast][0] * SS;
            maxpool_kernel<<<(tot + 255) / 256, 256, 0, stream>>>(
                src, out2, K, coutp[Llast], LC[Llast][0]);

            // Refine MLP + softmax offset: only last branch's result survives
            if (br == 2) {
                const bf16* rin = (const bf16*)(ws + offX);
                for (int j = 0; j < 11; ++j) {
                    int L = 9 + j;
                    bf16* o = bufs[j & 1];
                    dim3 g(Ncols / 512, coutp[L] / 16);
                    gemm_bn_relu_wmma<<<g, 256, 0, stream>>>(
                        rin, (const bf16*)(ws + offW[L]),
                        (const float*)(ws + offSc[L]), (const float*)(ws + offSh[L]),
                        o, Ncols, cinp[L], coutp[L]);
                    rin = o;
                }
                float* out1 = out + BB * 3 * SS + (size_t)b * 3 * SS;
                refine_offset_kernel<<<SS / 4, 128, 0, stream>>>(
                    rin, (const float*)(ws + offGx), out1);
            }
        }
    }
}
